// Pointer_Generator_85134841741887
// MI455X (gfx1250) — matrix operand order repaired and checked by reading, not compiled, not run on hardware
//
#include <hip/hip_runtime.h>
#include <hip/hip_bf16.h>
#include <math.h>

typedef __attribute__((ext_vector_type(16))) _Float16 v16h;
typedef __attribute__((ext_vector_type(8)))  float    v8f;
typedef __attribute__((ext_vector_type(4)))  float    v4f;

#define BB 4
#define TT 512
#define SS 512
#define DD 512
#define HH 8
#define VV 32000
#define EPSV 1e-10f

// ---------------------------------------------------------------------------
// Kernel 1: pointer/generator gate  pg = sigmoid(dec_output @ w + b)
// One wave per 16-row tile of [B*T, D]; K=512 as 16 x v_wmma_f32_16x16x32_f16.
// Every B column = the current 32-wide chunk of w, so all result columns hold
// the same dot product; lanes 0/16 (N=0) extract rows 0-7 / 8-15.
// ---------------------------------------------------------------------------
__global__ void __launch_bounds__(32)
pg_gate_wmma(const float* __restrict__ dec, const float* __restrict__ w,
             const float* __restrict__ bias, float* __restrict__ pg) {
  const int lane = threadIdx.x;          // 0..31
  const int m    = lane & 15;            // row within 16-row tile
  const int hi   = lane >> 4;            // half-wave selector
  const int row_base = blockIdx.x * 16;
  const float* arow = dec + (size_t)(row_base + m) * DD;

  v8f c = {};
  for (int k0 = 0; k0 < DD; k0 += 32) {
    v16h a, bm;
    // 16-bit A 16x32 layout: lanes 0-15 -> K = e (e<8), 16+e-8 (e>=8)
    //                        lanes 16-31 -> K = 8+e,   24+e-8
    #pragma unroll
    for (int e = 0; e < 8; ++e) {
      a[e]     = (_Float16)arow[k0 + hi * 8 + e];
      a[e + 8] = (_Float16)arow[k0 + 16 + hi * 8 + e];
    }
    // B 32x16: same K order as A (lanes 0-15 -> K=j / 16+j-8, lanes 16-31 -> 8+j / 24+j-8)
    #pragma unroll
    for (int j = 0; j < 16; ++j)
      bm[j] = (_Float16)w[k0 + hi * 8 + (j & 7) + (j >> 3) * 16];

    c = __builtin_amdgcn_wmma_f32_16x16x32_f16(
        /*neg_a=*/false, a, /*neg_b=*/false, bm,
        /*c_mod=*/(short)0, c, /*reuse_a=*/false, /*reuse_b=*/false);
  }

  // C/D layout: lane 0 -> N=0, M=r in c[r]; lane 16 -> N=0, M=8+r.
  if (m == 0) {
    const float bv = bias[0];
    #pragma unroll
    for (int r = 0; r < 8; ++r) {
      float x = c[r] + bv;
      pg[row_base + hi * 8 + r] = 1.0f / (1.0f + __expf(-x));
    }
  }
}

// ---------------------------------------------------------------------------
// Kernel 2 (fully fused, one 512-thread block per (b,t) row):
//   1) attention: head-mean over H=8 (16 KB/row), block softmax over S=512,
//      weighted by (1-pg)                              -> one value per thread
//   2) vocab softmax stats over V=32000 (float4 / global_load_b128 passes)
//   3) LDS scatter-add of attention mass onto vocab ids (ds_add_f32)
//   4) out = log(max(pg*softmax + scatter, EPS)), float4 non-temporal stores
// Dynamic LDS: 512-float reduction buffer + 32000-float scatter array
// (~130 KB -> exploits CDNA5's 320 KB WGP LDS; 2 blocks/WGP, 32 waves).
// ---------------------------------------------------------------------------
extern __shared__ float smem3[];
__global__ void __launch_bounds__(512)
fused_row(const float* __restrict__ fo, const float* __restrict__ attw,
          const int* __restrict__ enc, const float* __restrict__ pg,
          float* __restrict__ out) {
  float* red = smem3;          // [512]
  float* sc  = smem3 + 512;    // [VV]
  const int row = blockIdx.x;  // b*T + t
  const int b   = row / TT;
  const int t   = row - b * TT;
  const int tid = threadIdx.x;

  // zero the scatter array (visible via the reduction barriers below)
  for (int v = tid; v < VV; v += 512) sc[v] = 0.0f;

  // ---- attention: s = tid (S == 512, exactly one per thread) --------------
  const size_t abase = (size_t)b * HH * TT * SS + (size_t)t * SS;
  float av = 0.0f;
  #pragma unroll
  for (int h = 0; h < HH; ++h)
    av += attw[abase + (size_t)h * TT * SS + tid];
  av *= (1.0f / HH);

  red[tid] = av;
  __syncthreads();
  for (int s2 = 256; s2 > 0; s2 >>= 1) {
    if (tid < s2) red[tid] = fmaxf(red[tid], red[tid + s2]);
    __syncthreads();
  }
  const float amx = red[0];
  __syncthreads();
  const float ae = __expf(av - amx);
  red[tid] = ae;
  __syncthreads();
  for (int s2 = 256; s2 > 0; s2 >>= 1) {
    if (tid < s2) red[tid] += red[tid + s2];
    __syncthreads();
  }
  const float pgv    = pg[row];
  const float attval = (1.0f - pgv) * ae / red[0];
  __syncthreads();   // red[] about to be reused

  // ---- vocab softmax stats, vectorized (8000 float4 per row) --------------
  const v4f* frow4 = (const v4f*)(fo + (size_t)row * VV);
  float mx = -3.402823466e+38f;
  for (int i = tid; i < VV / 4; i += 512) {
    v4f x = frow4[i];
    mx = fmaxf(mx, fmaxf(fmaxf(x[0], x[1]), fmaxf(x[2], x[3])));
  }
  red[tid] = mx;
  __syncthreads();
  for (int s2 = 256; s2 > 0; s2 >>= 1) {
    if (tid < s2) red[tid] = fmaxf(red[tid], red[tid + s2]);
    __syncthreads();
  }
  mx = red[0];
  __syncthreads();

  float sum = 0.0f;
  for (int i = tid; i < VV / 4; i += 512) {
    v4f x = frow4[i];
    sum += __expf(x[0] - mx) + __expf(x[1] - mx) +
           __expf(x[2] - mx) + __expf(x[3] - mx);
  }
  red[tid] = sum;
  __syncthreads();
  for (int s2 = 256; s2 > 0; s2 >>= 1) {
    if (tid < s2) red[tid] += red[tid + s2];
    __syncthreads();
  }
  const float inv = 1.0f / red[0];

  // ---- scatter: one ds_add_f32 per source position ------------------------
  atomicAdd(&sc[enc[b * SS + tid]], attval);
  __syncthreads();

  // ---- fused combine + log, write-once non-temporal b128 stores -----------
  v4f* out4 = (v4f*)(out + (size_t)row * VV);
  const v4f* sc4 = (const v4f*)sc;
  for (int i = tid; i < VV / 4; i += 512) {
    v4f x  = frow4[i];
    v4f s4 = sc4[i];
    v4f r;
    #pragma unroll
    for (int c = 0; c < 4; ++c) {
      float val = pgv * __expf(x[c] - mx) * inv + s4[c];
      r[c] = __logf(fmaxf(val, EPSV));
    }
    __builtin_nontemporal_store(r, &out4[i]);
  }
}

// ---------------------------------------------------------------------------
extern "C" void kernel_launch(void* const* d_in, const int* in_sizes, int n_in,
                              void* d_out, int out_size, void* d_ws, size_t ws_size,
                              hipStream_t stream) {
  const float* dec  = (const float*)d_in[0];   // [B,T,D]
  const float* fo   = (const float*)d_in[1];   // [B,T,V]
  const float* attw = (const float*)d_in[2];   // [B,H,T,S]
  const int*   enc  = (const int*)  d_in[3];   // [B,S]
  const float* w    = (const float*)d_in[4];   // [D,1]
  const float* bias = (const float*)d_in[5];   // [1]
  float* out = (float*)d_out;

  float* pg = (float*)d_ws;                    // B*T floats (8 KB)

  // 1) gate via WMMA: 2048 rows / 16 rows-per-wave = 128 single-wave blocks
  pg_gate_wmma<<<(BB * TT) / 16, 32, 0, stream>>>(dec, w, bias, pg);

  // 2) fused attention-softmax + vocab-softmax + LDS scatter + log
  const size_t smem_bytes = (size_t)(512 + VV) * sizeof(float);
  fused_row<<<BB * TT, 512, smem_bytes, stream>>>(fo, attw, enc, pg, out);
}
